// Encoder_17798344474970
// MI455X (gfx1250) — matrix operand order repaired and checked by reading, not compiled, not run on hardware
//
#include <hip/hip_runtime.h>
#include <hip/hip_bf16.h>

// ---------------------------------------------------------------------------
// tanh-RNN encoder for MI455X (gfx1250, wave32).
//   Phase 1: x_proj = query @ W_ih^T + (b_ih + b_hh)   -> bf16 WMMA GEMM
//   Phase 2: h_t = tanh(x_proj_t + h_{t-1} @ W_hh^T)   -> 16-WG cluster scan
// Matrix core path: v_wmma_f32_16x16x32_bf16 (fp32 accumulate).
// Data movement: GLOBAL_LOAD_ASYNC_TO_LDS_B128 where no conversion is needed.
// ---------------------------------------------------------------------------

typedef __attribute__((ext_vector_type(16))) __bf16 v16bf;
typedef __attribute__((ext_vector_type(8)))  float  v8f;
typedef __attribute__((ext_vector_type(4)))  int    v4i;

#define RNN_B 64
#define RNN_T 2048
#define RNN_F 256
#define RNN_H 512

// float -> bf16 bits, round-to-nearest-even
__device__ __forceinline__ unsigned short f2bf(float f) {
    unsigned int u = __float_as_uint(f);
    u += 0x7FFFu + ((u >> 16) & 1u);
    return (unsigned short)(u >> 16);
}

// Packed fp32x2 -> bf16x2 (v_cvt_pk_bf16_f32 when available).
__device__ __forceinline__ void store_bf16_pair(unsigned short* dst, float lo, float hi) {
#if __has_builtin(__builtin_amdgcn_cvt_pk_bf16_f32)
    typedef __attribute__((ext_vector_type(2))) __bf16 v2bf;
    union { v2bf v; unsigned int u; } p;
    p.v = __builtin_amdgcn_cvt_pk_bf16_f32(lo, hi);
    *reinterpret_cast<unsigned int*>(dst) = p.u;
#else
    dst[0] = f2bf(lo);
    dst[1] = f2bf(hi);
#endif
}

// ---- Async global->LDS DMA (gfx1250), with safe fallback -------------------
// Builtin signature (from hipcc diagnostics): arg0 = v4i in AS(1) (global),
// arg1 = LDS destination, then imm offset + imm cpol.
#if __has_builtin(__builtin_amdgcn_global_load_async_to_lds_b128)
#define HAVE_ASYNC_LDS 1
typedef __attribute__((address_space(1))) v4i g_v4i;
typedef __attribute__((address_space(3))) v4i l_v4i;
__device__ __forceinline__ void async_copy_b128(const void* gsrc, void* ldst) {
    __builtin_amdgcn_global_load_async_to_lds_b128(
        (g_v4i*)(void*)gsrc,
        (l_v4i*)ldst,
        /*imm offset=*/0, /*cpol=*/0);
}
__device__ __forceinline__ void async_wait0() {
#if __has_builtin(__builtin_amdgcn_s_wait_asynccnt)
    __builtin_amdgcn_s_wait_asynccnt(0);
#else
    asm volatile("s_wait_asynccnt 0x0" ::: "memory");
#endif
}
#else
#define HAVE_ASYNC_LDS 0
#endif

union FragU {
    v16bf v;
    uint4 q[2];
};

// A-matrix 16x32 bf16 fragment (ISA 7.12.2):
// lane<16 : row=lane,    K = {0..7}   in elems 0-7,  {16..23} in elems 8-15
// lane>=16: row=lane-16, K = {8..15}  in elems 0-7,  {24..31} in elems 8-15
__device__ __forceinline__ v16bf load_a_frag(const unsigned short* base, int lane,
                                             int row_stride) {
    const unsigned short* p = base + (size_t)(lane & 15) * row_stride
                                   + ((lane & 16) ? 8 : 0);
    FragU f;
    f.q[0] = *reinterpret_cast<const uint4*>(p);        // K chunk +0
    f.q[1] = *reinterpret_cast<const uint4*>(p + 16);   // K chunk +16
    return f.v;
}

// B-matrix 32x16 bf16 fragment. B = W^T, so column n of B == row n of W
// (contiguous in K). lane<16: col=lane, K={0..7,16..23}; lane>=16: col=lane-16, K={8..15,24..31}.
__device__ __forceinline__ v16bf load_b_frag(const unsigned short* base, int lane,
                                             int row_stride) {
    const unsigned short* p = base + (size_t)(lane & 15) * row_stride
                                   + ((lane & 16) ? 8 : 0);
    FragU f;
    f.q[0] = *reinterpret_cast<const uint4*>(p);        // K chunk +0
    f.q[1] = *reinterpret_cast<const uint4*>(p + 16);   // K chunk +16
    return f.v;
}

__device__ __forceinline__ v8f zero8() {
    v8f z = {0.f, 0.f, 0.f, 0.f, 0.f, 0.f, 0.f, 0.f};
    return z;
}

__device__ __forceinline__ v8f wmma_bf16(v16bf a, v16bf b, v8f c) {
    return __builtin_amdgcn_wmma_f32_16x16x32_bf16(
        /*neg_a=*/false, a, /*neg_b=*/false, b,
        /*c_mod=*/(short)0, c, /*reuse_a=*/false, /*reuse_b=*/false);
}

// ---------------------------------------------------------------------------
// Prep: convert weights to bf16, fuse biases, zero h0 ping-pong buffers.
// ---------------------------------------------------------------------------
__global__ __launch_bounds__(256) void rnn_prep(
        const float* __restrict__ Wih, const float* __restrict__ Whh,
        const float* __restrict__ bih, const float* __restrict__ bhh,
        unsigned short* __restrict__ wih_bf, unsigned short* __restrict__ whh_bf,
        float* __restrict__ bias, unsigned short* __restrict__ hbuf) {
    int i = blockIdx.x * blockDim.x + threadIdx.x;
    if (i < RNN_H * RNN_F) wih_bf[i] = f2bf(Wih[i]);
    if (i < RNN_H * RNN_H) whh_bf[i] = f2bf(Whh[i]);
    if (i < RNN_H)         bias[i]   = bih[i] + bhh[i];
    if (i < 2 * RNN_B * RNN_H) hbuf[i] = 0;   // bf16 +0.0
}

// ---------------------------------------------------------------------------
// Phase 1: x_proj[m, n] = sum_k query[m,k] * W_ih[n,k] + bias[n]
// M = B*T = 131072, N = 512, K = 256. Block tile 128x128, K-chunk 32.
// 8 waves / block, each wave owns a 32x64 subtile (2x4 WMMA accumulators).
// ---------------------------------------------------------------------------
#define G1_BM 128
#define G1_BN 128
#define G1_BK 32

__global__ __launch_bounds__(256) void xproj_gemm(
        const float* __restrict__ X,              // [M, 256] fp32
        const unsigned short* __restrict__ Wb,    // [512, 256] bf16
        const float* __restrict__ bias,           // [512]
        float* __restrict__ out) {                // [M, 512] fp32
    __shared__ unsigned short sA[G1_BM * G1_BK];  // 8 KB
    __shared__ unsigned short sB[G1_BN * G1_BK];  // 8 KB

    const int N = RNN_H, K = RNN_F;
    const int ntn = N / G1_BN;                    // 4
    const int bm = (blockIdx.x / ntn) * G1_BM;
    const int bn = (blockIdx.x % ntn) * G1_BN;

    const int tid  = threadIdx.x;
    const int wid  = tid >> 5;
    const int lane = tid & 31;
    const int wm = (wid >> 1) * 32;               // 0,32,64,96
    const int wn = (wid & 1) * 64;                // 0,64

    v8f acc[2][4];
#pragma unroll
    for (int i = 0; i < 2; ++i)
#pragma unroll
        for (int j = 0; j < 4; ++j) acc[i][j] = zero8();

    const int r = tid >> 1;                       // 0..127
    const int c = (tid & 1) * 16;                 // 0 or 16

    for (int k0 = 0; k0 < K; k0 += G1_BK) {
        // Stage B tile first via async DMA (bf16 weights, no conversion):
        // 128x32 bf16 = 512 x 16B chunks, 2 per thread.
#if HAVE_ASYNC_LDS
#pragma unroll
        for (int cc = 0; cc < 2; ++cc) {
            int ch  = tid * 2 + cc;               // 0..511
            int row = ch >> 2;                    // 0..127
            int col = (ch & 3) * 8;               // 0,8,16,24
            async_copy_b128(Wb + (size_t)(bn + row) * K + k0 + col,
                            &sB[row * G1_BK + col]);
        }
#else
        {
            const unsigned short* bsrc = Wb + (size_t)(bn + r) * K + k0 + c;
#pragma unroll
            for (int j = 0; j < 16; ++j) sB[r * G1_BK + c + j] = bsrc[j];
        }
#endif
        // Stage A tile (fp32 -> bf16 on the fly): 128x32
        const float* asrc = X + (size_t)(bm + r) * K + k0 + c;
#pragma unroll
        for (int j = 0; j < 16; j += 2)
            store_bf16_pair(&sA[r * G1_BK + c + j], asrc[j], asrc[j + 1]);
#if HAVE_ASYNC_LDS
        async_wait0();
#endif
        __syncthreads();

#pragma unroll
        for (int i = 0; i < 2; ++i) {
            v16bf a = load_a_frag(sA + (wm + i * 16) * G1_BK, lane, G1_BK);
#pragma unroll
            for (int j = 0; j < 4; ++j) {
                v16bf b = load_b_frag(sB + (wn + j * 16) * G1_BK, lane, G1_BK);
                acc[i][j] = wmma_bf16(a, b, acc[i][j]);
            }
        }
        __syncthreads();
    }

    // Epilogue: C/D layout -> fp32 store with bias.
#pragma unroll
    for (int i = 0; i < 2; ++i) {
#pragma unroll
        for (int j = 0; j < 4; ++j) {
#pragma unroll
            for (int v = 0; v < 8; ++v) {
                int row = bm + wm + i * 16 + v + ((lane & 16) ? 8 : 0);
                int col = bn + wn + j * 16 + (lane & 15);
                out[(size_t)row * N + col] = acc[i][j][v] + bias[col];
            }
        }
    }
}

// ---------------------------------------------------------------------------
// Phase 2: recurrent scan. Launched as a 16-workgroup cluster (one WG per WGP
// on a shader engine). Each WG owns a 32-column slice of H; its W_hh slice
// (32x512 bf16 = 32 KB) lives in LDS for the whole scan. h_t ping-pongs as
// bf16 [64x512] through global/L2. Cluster split-barrier syncs each step.
// ---------------------------------------------------------------------------
__global__ __launch_bounds__(256) void rnn_scan(
        const float* __restrict__ xproj,          // [B*T, 512] fp32, row b*T+t
        const unsigned short* __restrict__ whh_bf,// [512, 512] bf16
        unsigned short* __restrict__ hbuf,        // 2 x [64*512] bf16 ping-pong
        float* __restrict__ output,               // [B, T, 512] fp32
        float* __restrict__ hidden) {             // [B, 512] fp32
    __shared__ unsigned short sW[32 * RNN_H];     // 32 KB: rows n0..n0+31 of W_hh

    const int n0   = blockIdx.x * 32;             // this WG's H-slice
    const int tid  = threadIdx.x;
    const int wid  = tid >> 5;
    const int lane = tid & 31;
    const int m0    = (wid >> 1) * 16;            // batch-row tile: 0,16,32,48
    const int lcol0 = (wid & 1) * 16;             // local col tile: 0 or 16

    // Load the time-invariant W_hh slice once (contiguous rows) via async DMA.
    const unsigned short* wsrc = whh_bf + (size_t)n0 * RNN_H;
#if HAVE_ASYNC_LDS
    for (int ch = tid; ch < 2048; ch += 256)      // 2048 x 16B chunks
        async_copy_b128(wsrc + ch * 8, &sW[ch * 8]);
    async_wait0();
#else
    for (int i = tid * 64, e = i + 64; i < e; ++i) sW[i] = wsrc[i];
#endif
    __syncthreads();

    unsigned short* h_read  = hbuf;
    unsigned short* h_write = hbuf + RNN_B * RNN_H;

    for (int t = 0; t < RNN_T; ++t) {
        // Prefetch next step's x_proj elements for this lane (global_prefetch_b8).
        if (t + 1 < RNN_T) {
            const float* pf = xproj + ((size_t)(m0 + (lane & 15)) * RNN_T + t + 1) * RNN_H
                                    + n0 + lcol0;
            __builtin_prefetch(pf, 0, 1);
        }

        v8f acc = zero8();
#pragma unroll 4
        for (int kc = 0; kc < RNN_H; kc += 32) {
            v16bf a = load_a_frag(h_read + (size_t)m0 * RNN_H + kc, lane, RNN_H);
            v16bf b = load_b_frag(sW + (size_t)lcol0 * RNN_H + kc, lane, RNN_H);
            acc = wmma_bf16(a, b, acc);
        }

        // Epilogue: add x_proj, tanh, write output + next-h (+ final hidden).
#pragma unroll
        for (int v = 0; v < 8; ++v) {
            int brow = m0 + v + ((lane & 16) ? 8 : 0);       // batch index
            int gcol = n0 + lcol0 + (lane & 15);             // hidden index
            size_t orow = (size_t)brow * RNN_T + t;
            float x = acc[v] + xproj[orow * RNN_H + gcol];
            float y = tanhf(x);
            output[orow * RNN_H + gcol] = y;
            h_write[(size_t)brow * RNN_H + gcol] = f2bf(y);
            if (t == RNN_T - 1) hidden[(size_t)brow * RNN_H + gcol] = y;
        }

        // Make h_write visible device-wide, then sync the whole cluster.
        __threadfence();
        __syncthreads();
        __builtin_amdgcn_s_cluster_barrier();

        unsigned short* tmp = h_read; h_read = h_write; h_write = tmp;
    }
}

// ---------------------------------------------------------------------------
extern "C" void kernel_launch(void* const* d_in, const int* in_sizes, int n_in,
                              void* d_out, int out_size, void* d_ws, size_t ws_size,
                              hipStream_t stream) {
    const float* query = (const float*)d_in[0];   // [64, 2048, 256]
    const float* W_ih  = (const float*)d_in[1];   // [512, 256]
    const float* W_hh  = (const float*)d_in[2];   // [512, 512]
    const float* b_ih  = (const float*)d_in[3];   // [512]
    const float* b_hh  = (const float*)d_in[4];   // [512]

    float* out    = (float*)d_out;                         // output [64,2048,512]
    float* hidden = out + (size_t)RNN_B * RNN_T * RNN_H;   // then hidden [1,64,512]

    char* ws = (char*)d_ws;
    unsigned short* wih_bf = (unsigned short*)(ws + 0);            // 256 KB
    unsigned short* whh_bf = (unsigned short*)(ws + (256 << 10));  // 512 KB
    float*          bias   = (float*)(ws + (768 << 10));           // 2 KB
    unsigned short* hbuf   = (unsigned short*)(ws + (772 << 10));  // 128 KB
    float*          xproj  = (float*)(ws + (1 << 20));             // 256 MB

    rnn_prep<<<(RNN_H * RNN_H + 255) / 256, 256, 0, stream>>>(
        W_ih, W_hh, b_ih, b_hh, wih_bf, whh_bf, bias, hbuf);

    const int M = RNN_B * RNN_T;
    const int nblocks = (M / G1_BM) * (RNN_H / G1_BN);   // 1024 * 4 = 4096
    xproj_gemm<<<nblocks, 256, 0, stream>>>(query, wih_bf, bias, xproj);

    // 16 workgroups = one cluster spanning 16 WGPs; 32 H-columns each.
    rnn_scan<<<16, 256, 0, stream>>>(xproj, whh_bf, hbuf, out, hidden);
}
